// ProbSparseAttention_49881750175904
// MI455X (gfx1250) — matrix operand order repaired and checked by reading, not compiled, not run on hardware
//
#include <hip/hip_runtime.h>

// Problem constants (match reference setup_inputs)
#define BB    8
#define LL    2048
#define DMD   512
#define HH    8
#define SS    40
#define NTOP  40

typedef __attribute__((ext_vector_type(16))) __bf16 v16bf;
typedef __attribute__((ext_vector_type(8)))  float  v8f;

union WFrag {
    v16bf v;
    uint4 q[2];
};

__device__ __forceinline__ unsigned short f2bf(float f) {
    unsigned int x = __float_as_uint(f);
    unsigned int r = x + 0x7FFFu + ((x >> 16) & 1u); // round-to-nearest-even
    return (unsigned short)(r >> 16);
}
__device__ __forceinline__ unsigned int pack_bf2(float a, float b) {
    return (unsigned int)f2bf(a) | ((unsigned int)f2bf(b) << 16);
}

// ---------------------------------------------------------------------------
// 0) one-shot f32 -> bf16 conversion prepass (RNE), float4 -> uint2
// ---------------------------------------------------------------------------
__global__ __launch_bounds__(256) void cvt_bf16_kernel(const float* __restrict__ src,
                                                       unsigned short* __restrict__ dst) {
    size_t i = (size_t)blockIdx.x * 256 + threadIdx.x;   // float4 index
    float4 v = ((const float4*)src)[i];
    uint2 p;
    p.x = pack_bf2(v.x, v.y);
    p.y = pack_bf2(v.z, v.w);
    ((uint2*)dst)[i] = p;
}

// ---------------------------------------------------------------------------
// 1) vmean[b][d] = mean over L of values[b][l][d]
// ---------------------------------------------------------------------------
__global__ __launch_bounds__(DMD) void vmean_kernel(const float* __restrict__ V,
                                                    float* __restrict__ vmean) {
    int b = blockIdx.x;
    int d = threadIdx.x;
    const float* p = V + (size_t)b * LL * DMD + d;
    float s = 0.f;
    for (int l = 0; l < LL; ++l) s += p[(size_t)l * DMD];
    vmean[b * DMD + d] = s * (1.0f / (float)LL);
}

// ---------------------------------------------------------------------------
// 2) rows[b] = ((vmean[b]) @ Wv^T + bv) @ Wo^T + bo
// ---------------------------------------------------------------------------
__global__ __launch_bounds__(DMD) void outrow_kernel(const float* __restrict__ vmean,
                                                     const float* __restrict__ Wv,
                                                     const float* __restrict__ bv,
                                                     const float* __restrict__ Wo,
                                                     const float* __restrict__ bo,
                                                     float* __restrict__ rows) {
    __shared__ float ctx[DMD];
    int b = blockIdx.x;
    int c = threadIdx.x;
    const float* vm = vmean + b * DMD;
    float s = bv[c];
    const float* wr = Wv + (size_t)c * DMD;
    for (int j = 0; j < DMD; ++j) s += vm[j] * wr[j];
    ctx[c] = s;
    __syncthreads();
    float o = bo[c];
    const float* orow = Wo + (size_t)c * DMD;
    for (int j = 0; j < DMD; ++j) o += ctx[j] * orow[j];
    rows[b * DMD + c] = o;
}

// ---------------------------------------------------------------------------
// 3) out[b][l][:] = rows[b][:]   (float4 broadcast store)
// ---------------------------------------------------------------------------
__global__ __launch_bounds__(256) void bcast_kernel(const float* __restrict__ rows,
                                                    float* __restrict__ out) {
    size_t i = (size_t)blockIdx.x * 256 + threadIdx.x;
    size_t flat = i * 4;
    int d = (int)(flat % DMD);
    int b = (int)(flat / ((size_t)LL * DMD));
    float4 v = *(const float4*)(rows + b * DMD + d);
    *(float4*)(out + flat) = v;
}

// ---------------------------------------------------------------------------
// 4) WMMA projection:  Y = Xbf @ Wbf^T + bias   (bf16 in, f32 out)
//    64x128 output tile / block; 256 threads = 8 waves; 4 accumulators / wave.
//    LDS staged via global_load_async_to_lds_b128 (ASYNCcnt), fragments read
//    as two aligned ds_load_b128 per matrix per the ISA 16-bit layouts.
// ---------------------------------------------------------------------------
#define TMP 64
#define TNP 128
#define TKP 64
#define ASTR 72   // LDS row stride in shorts (144 B = 9 x 16 B, bank-skewed)

__global__ __launch_bounds__(256) void proj_wmma_kernel(const unsigned short* __restrict__ Xbf,
                                                        const unsigned short* __restrict__ Wbf,
                                                        const float* __restrict__ bias,
                                                        float* __restrict__ Y) {
    __shared__ unsigned short As[TMP][ASTR];   // As[m][k]
    __shared__ unsigned short Bs[TNP][ASTR];   // Bs[n][k] = W[col0+n][k0+k]

    const int row0 = blockIdx.x * TMP;
    const int col0 = blockIdx.y * TNP;
    const int tid  = threadIdx.x;
    const int lane = tid & 31;
    const int wv   = tid >> 5;
    const int mg   = wv & 3;        // m sub-tile (0..3)
    const int nset = wv >> 2;       // n half: groups nset*4 .. nset*4+3

    const unsigned long long xbase = (unsigned long long)Xbf;
    const unsigned long long wbase = (unsigned long long)Wbf;
    const unsigned asBase = (unsigned)(size_t)&As[0][0];
    const unsigned bsBase = (unsigned)(size_t)&Bs[0][0];

    v8f acc[4];
    #pragma unroll
    for (int j = 0; j < 4; ++j) acc[j] = (v8f){0.f,0.f,0.f,0.f,0.f,0.f,0.f,0.f};

    for (int k0 = 0; k0 < DMD; k0 += TKP) {
        // ---- async-stage A (64x64 bf16) and B (128x64 bf16) tiles into LDS
        #pragma unroll
        for (int i = 0; i < 2; ++i) {                    // A: 512 16B-chunks
            int idx = tid + i * 256;
            int r = idx >> 3, c = idx & 7;
            unsigned voff = ((unsigned)(row0 + r) * DMD + (unsigned)k0) * 2u + (unsigned)c * 16u;
            unsigned laddr = asBase + (unsigned)r * (ASTR * 2) + (unsigned)c * 16u;
            asm volatile("global_load_async_to_lds_b128 %0, %1, %2"
                         :: "v"(laddr), "v"(voff), "s"(xbase) : "memory");
        }
        #pragma unroll
        for (int i = 0; i < 4; ++i) {                    // B: 1024 16B-chunks
            int idx = tid + i * 256;
            int r = idx >> 3, c = idx & 7;
            unsigned voff = ((unsigned)(col0 + r) * DMD + (unsigned)k0) * 2u + (unsigned)c * 16u;
            unsigned laddr = bsBase + (unsigned)r * (ASTR * 2) + (unsigned)c * 16u;
            asm volatile("global_load_async_to_lds_b128 %0, %1, %2"
                         :: "v"(laddr), "v"(voff), "s"(wbase) : "memory");
        }
        asm volatile("s_wait_asynccnt 0" ::: "memory");
        __syncthreads();

        // ---- two K=32 WMMA steps per 64-wide chunk
        #pragma unroll
        for (int kk = 0; kk < TKP; kk += 32) {
            // A fragment (16x32 bf16): lanes 0-15 K-halves 0..7/16..23,
            // lanes 16-31 K-halves 8..15/24..31 -> two aligned b128 reads
            WFrag a;
            const int mrow  = mg * 16 + (lane & 15);
            const int abase = (lane < 16) ? 0 : 8;
            a.q[0] = *(const uint4*)&As[mrow][kk + abase];
            a.q[1] = *(const uint4*)&As[mrow][kk + 16 + abase];

            const int bbase = (lane < 16) ? 0 : 8;
            #pragma unroll
            for (int j = 0; j < 4; ++j) {
                WFrag b;
                const int nrow = (nset * 4 + j) * 16 + (lane & 15);
                b.q[0] = *(const uint4*)&Bs[nrow][kk + bbase];
                b.q[1] = *(const uint4*)&Bs[nrow][kk + 16 + bbase];
                acc[j] = __builtin_amdgcn_wmma_f32_16x16x32_bf16(
                             false, a.v, false, b.v, (short)0, acc[j], false, false);
            }
        }
        __syncthreads();
    }

    // ---- epilogue: C/D layout: n = lane&15; VGPR r -> m = r + (lane<16?0:8)
    const int n  = lane & 15;
    const int mb = (lane < 16) ? 0 : 8;
    #pragma unroll
    for (int j = 0; j < 4; ++j) {
        const int gc = col0 + (nset * 4 + j) * 16 + n;
        const float bbv = bias[gc];
        #pragma unroll
        for (int r = 0; r < 8; ++r) {
            int grow = row0 + mg * 16 + mb + r;
            Y[(size_t)grow * DMD + gc] = acc[j][r] + bbv;
        }
    }
}

// ---------------------------------------------------------------------------
// 5) score: M[b,h,l] = max_s(q.k_s) - mean_s(q.k_s), s over 40 samples
// ---------------------------------------------------------------------------
__global__ __launch_bounds__(HH * SS) void score_kernel(const float* __restrict__ Q,
                                                        const float* __restrict__ K,
                                                        const int* __restrict__ samp,
                                                        float* __restrict__ M) {
    __shared__ float qrow[DMD];
    __shared__ float dots[HH][SS];
    const int b = blockIdx.x / LL;
    const int l = blockIdx.x % LL;
    const int tid = threadIdx.x;

    for (int i = tid; i < DMD; i += HH * SS)
        qrow[i] = Q[(size_t)(b * LL + l) * DMD + i];
    __syncthreads();

    {
        const int h = tid / SS;
        const int s = tid % SS;
        const int ki = samp[l * SS + s];
        const float* krow = K + (size_t)(b * LL + ki) * DMD + h * 64;
        __builtin_prefetch(krow, 0, 1);  // global_prefetch_b8
        const float* qh = qrow + h * 64;
        float acc = 0.f;
        #pragma unroll
        for (int d = 0; d < 64; ++d) acc += qh[d] * krow[d];
        dots[h][s] = acc;
    }
    __syncthreads();

    if (tid < HH) {
        float mx = dots[tid][0];
        float sm = 0.f;
        #pragma unroll
        for (int s = 0; s < SS; ++s) {
            float v = dots[tid][s];
            mx = fmaxf(mx, v);
            sm += v;
        }
        M[((size_t)b * HH + tid) * LL + l] = mx - sm * (1.0f / (float)SS);
    }
}

// ---------------------------------------------------------------------------
// 6) topk: per (b,h) select top-40 indices of M over L (stable, tie->lower idx)
// ---------------------------------------------------------------------------
__global__ __launch_bounds__(256) void topk_kernel(float* __restrict__ M,
                                                   float* __restrict__ out_idx) {
    __shared__ float sv[256];
    __shared__ int   si[256];
    float* m = M + (size_t)blockIdx.x * LL;
    const int tid = threadIdx.x;

    for (int t = 0; t < NTOP; ++t) {
        float best = -3.4e38f;
        int   bi   = LL;
        for (int i = tid; i < LL; i += 256) {
            float v = m[i];
            if (v > best) { best = v; bi = i; }
        }
        sv[tid] = best; si[tid] = bi;
        __syncthreads();
        for (int off = 128; off > 0; off >>= 1) {
            if (tid < off) {
                float ov = sv[tid + off]; int oi = si[tid + off];
                if (ov > sv[tid] || (ov == sv[tid] && oi < si[tid])) {
                    sv[tid] = ov; si[tid] = oi;
                }
            }
            __syncthreads();
        }
        if (tid == 0) {
            out_idx[(size_t)blockIdx.x * NTOP + t] = (float)si[0];
            m[si[0]] = -3.4e38f;
        }
        __syncthreads();
    }
}

// ---------------------------------------------------------------------------
extern "C" void kernel_launch(void* const* d_in, const int* in_sizes, int n_in,
                              void* d_out, int out_size, void* d_ws, size_t ws_size,
                              hipStream_t stream) {
    const float* queries = (const float*)d_in[0];
    const float* keys    = (const float*)d_in[1];
    const float* values  = (const float*)d_in[2];
    const float* Wq      = (const float*)d_in[3];
    const float* bq      = (const float*)d_in[4];
    const float* Wk      = (const float*)d_in[5];
    const float* bk      = (const float*)d_in[6];
    const float* Wv      = (const float*)d_in[7];
    const float* bv      = (const float*)d_in[8];
    const float* Wo      = (const float*)d_in[9];
    const float* bo      = (const float*)d_in[10];
    const int*   samp    = (const int*)d_in[11];

    float* out  = (float*)d_out;
    float* mtop = out + (size_t)BB * LL * DMD;     // tail of d_out: (B,H,NTOP)

    // workspace carve-up (f32 region, then bf16 region)
    float* Qws   = (float*)d_ws;
    float* Kws   = Qws + (size_t)BB * LL * DMD;
    float* Mws   = Kws + (size_t)BB * LL * DMD;
    float* vmean = Mws + (size_t)BB * HH * LL;
    float* rows  = vmean + BB * DMD;
    unsigned short* Xqbf = (unsigned short*)(rows + BB * DMD);
    unsigned short* Xkbf = Xqbf + (size_t)BB * LL * DMD;
    unsigned short* Wqbf = Xkbf + (size_t)BB * LL * DMD;
    unsigned short* Wkbf = Wqbf + (size_t)DMD * DMD;

    // 0) bf16 prepass for GEMM operands
    {
        unsigned nx4 = (unsigned)((size_t)BB * LL * DMD / 4);   // 2,097,152
        unsigned nw4 = (unsigned)((size_t)DMD * DMD / 4);       // 65,536
        cvt_bf16_kernel<<<nx4 / 256, 256, 0, stream>>>(queries, Xqbf);
        cvt_bf16_kernel<<<nx4 / 256, 256, 0, stream>>>(keys,    Xkbf);
        cvt_bf16_kernel<<<nw4 / 256, 256, 0, stream>>>(Wq,      Wqbf);
        cvt_bf16_kernel<<<nw4 / 256, 256, 0, stream>>>(Wk,      Wkbf);
    }
    // 1) mean of values over L
    vmean_kernel<<<BB, DMD, 0, stream>>>(values, vmean);
    // 2) per-batch output row
    outrow_kernel<<<BB, DMD, 0, stream>>>(vmean, Wv, bv, Wo, bo, rows);
    // 3) broadcast into out
    {
        size_t nf4 = (size_t)BB * LL * DMD / 4;
        bcast_kernel<<<(unsigned)(nf4 / 256), 256, 0, stream>>>(rows, out);
    }
    // 4) Q and K projections (WMMA, async-LDS staged)
    {
        dim3 grid((BB * LL) / TMP, DMD / TNP);
        proj_wmma_kernel<<<grid, 256, 0, stream>>>(Xqbf, Wqbf, bq, Qws);
        proj_wmma_kernel<<<grid, 256, 0, stream>>>(Xkbf, Wkbf, bk, Kws);
    }
    // 5) sparse sample scores
    score_kernel<<<BB * LL, HH * SS, 0, stream>>>(Qws, Kws, samp, Mws);
    // 6) top-k indices
    topk_kernel<<<BB * HH, 256, 0, stream>>>(Mws, mtop);
}
